// ReformerEncoderLayer_30872224923773
// MI455X (gfx1250) — matrix operand order repaired and checked by reading, not compiled, not run on hardware
//
#include <hip/hip_runtime.h>
#include <hip/hip_bf16.h>
#include <math.h>

// ---------------- problem constants ----------------
#define B_SZ   4
#define T_SZ   4096
#define D_SZ   512
#define H_SZ   8
#define DH_SZ  64
#define NHASH  4
#define NBUCK  64          // T / BUCKET
#define BUCKET 64
#define CHUNKS 256         // NHASH * NBUCK
#define BH_SZ  32          // B * H
#define SORTN  16384       // NHASH * T
#define MTOT   16384       // B * T

typedef __attribute__((ext_vector_type(16))) _Float16 v16h;
typedef __attribute__((ext_vector_type(4)))  _Float16 v4h;
typedef __attribute__((ext_vector_type(8)))  float    v8f;

// ---------------- WMMA helpers (CDNA5 16x16x32 f16, fp32 acc) ----------------
__device__ __forceinline__ v8f wmma_f16(v16h a, v16h b, v8f c) {
  // 8 args: (neg_a, A, neg_b, B, c_mod, C, reuse_a, reuse_b)
  return __builtin_amdgcn_wmma_f32_16x16x32_f16(false, a, false, b, (short)0, c,
                                                false, false);
}

__device__ __forceinline__ v8f zero8() {
  v8f z = {0.f, 0.f, 0.f, 0.f, 0.f, 0.f, 0.f, 0.f};
  return z;
}

__device__ __forceinline__ v4h cvt4(float4 f) {
  v4h o;
  o[0] = (_Float16)f.x; o[1] = (_Float16)f.y;
  o[2] = (_Float16)f.z; o[3] = (_Float16)f.w;
  return o;
}

// A fragment (16x32, f16), tile stored row-major (M x K), base -> [row0][k0].
// Layout: lanes 0-15 rows M=0..15 hold K 0..15;
//         lanes 16-31 rows M=0..15 hold K 16..31.
__device__ __forceinline__ v16h frag_a_rowmajor(const _Float16* base, int ld) {
  const int lane = threadIdx.x & 31;
  const _Float16* p = base + (lane & 15) * ld + ((lane >> 4) << 4);
  v16h f;
#pragma unroll
  for (int e = 0; e < 16; ++e) f[e] = p[e];
  return f;
}

// B fragment (32x16, f16) from an N-major tile: tile[n*ld + k], base -> [n0][k0].
// lanes 0-15 -> K=0..15 of column N=lane; lanes 16-31 -> K=16..31 of column N=lane-16.
__device__ __forceinline__ v16h frag_b_nmajor(const _Float16* base, int ld) {
  const int lane = threadIdx.x & 31;
  const _Float16* p = base + (lane & 15) * ld + ((lane >> 4) << 4);
  v16h f;
#pragma unroll
  for (int e = 0; e < 16; ++e) f[e] = p[e];
  return f;
}

// B fragment (32x16, f16) from a K-major tile: tile[k*ld + n], base -> [k0][n0].
__device__ __forceinline__ v16h frag_b_kmajor(const _Float16* base, int ld) {
  const int lane = threadIdx.x & 31;
  const _Float16* p = base + (((lane >> 4) << 4)) * ld + (lane & 15);
  v16h f;
#pragma unroll
  for (int e = 0; e < 16; ++e) f[e] = p[e * ld];
  return f;
}

// ---------------- generic WMMA GEMM: C = op(A @ W [+bias]) [+resid] ----------
// A: (M,K) f32 row-major, W: (K,N) f32 row-major. Block tile 128x128x64,
// 256 threads = 8 waves, each wave owns a 32x64 output sub-tile.
// Staging is two-phase: issue all 16 global b128 loads (deep LOADcnt pipeline),
// then convert + store to LDS.
__global__ __launch_bounds__(256) void gemm_f16wmma(
    const float* __restrict__ A, const float* __restrict__ W,
    const float* __restrict__ bias, const float* __restrict__ resid,
    float* __restrict__ C, int M, int N, int K, int do_relu) {
  __shared__ _Float16 As[128 * 64];   // row-major [m][k]
  __shared__ _Float16 Bs[128 * 64];   // N-major   [n][k]

  const int tid  = threadIdx.x;
  const int wid  = tid >> 5;
  const int wm   = wid & 3;          // 4 wave-rows  -> 32 rows each
  const int wn   = wid >> 2;         // 2 wave-cols  -> 64 cols each
  const int row0 = blockIdx.y * 128;
  const int col0 = blockIdx.x * 128;

  v8f acc[2][4];
#pragma unroll
  for (int i = 0; i < 2; ++i)
#pragma unroll
    for (int j = 0; j < 4; ++j) acc[i][j] = zero8();

  for (int k0 = 0; k0 < K; k0 += 64) {
    __syncthreads();
    // ---- phase 1: issue all staging loads back-to-back ----
    float4 ra[8], rw[8];
#pragma unroll
    for (int it = 0; it < 8; ++it) {                   // A tile 128x64
      int l = tid * 4 + it * 1024;
      int r = l >> 6, kk = l & 63;
      ra[it] = *(const float4*)(A + (size_t)(row0 + r) * K + k0 + kk);
    }
#pragma unroll
    for (int it = 0; it < 8; ++it) {                   // W tile 64x128
      int l = tid * 4 + it * 1024;
      int kk = l >> 7, n = l & 127;                    // l = kk*128 + n
      rw[it] = *(const float4*)(W + (size_t)(k0 + kk) * N + col0 + n);
    }
    if (k0 + 64 < K) {                                 // warm next K tile
      int r = tid >> 1, half = (tid & 1) << 5;
      __builtin_prefetch(A + (size_t)(row0 + r) * K + k0 + 64 + half, 0, 1);
      __builtin_prefetch(W + (size_t)(k0 + 64 + (tid >> 2)) * N + col0 + ((tid & 3) << 5), 0, 1);
    }
    // ---- phase 2: convert + store to LDS ----
#pragma unroll
    for (int it = 0; it < 8; ++it) {
      int l = tid * 4 + it * 1024;
      *(v4h*)(As + l) = cvt4(ra[it]);
    }
#pragma unroll
    for (int it = 0; it < 8; ++it) {
      int l = tid * 4 + it * 1024;
      int kk = l >> 7, n = l & 127;
      Bs[(n + 0) * 64 + kk] = (_Float16)rw[it].x;      // store N-major
      Bs[(n + 1) * 64 + kk] = (_Float16)rw[it].y;
      Bs[(n + 2) * 64 + kk] = (_Float16)rw[it].z;
      Bs[(n + 3) * 64 + kk] = (_Float16)rw[it].w;
    }
    __syncthreads();

#pragma unroll
    for (int ks = 0; ks < 64; ks += 32) {
      v16h af0 = frag_a_rowmajor(As + (wm * 32 + 0) * 64 + ks, 64);
      v16h af1 = frag_a_rowmajor(As + (wm * 32 + 16) * 64 + ks, 64);
#pragma unroll
      for (int nt = 0; nt < 4; ++nt) {
        v16h bf = frag_b_nmajor(Bs + (wn * 64 + nt * 16) * 64 + ks, 64);
        acc[0][nt] = wmma_f16(af0, bf, acc[0][nt]);
        acc[1][nt] = wmma_f16(af1, bf, acc[1][nt]);
      }
    }
  }

  const int lane = tid & 31;
#pragma unroll
  for (int mt = 0; mt < 2; ++mt) {
#pragma unroll
    for (int nt = 0; nt < 4; ++nt) {
#pragma unroll
      for (int r = 0; r < 8; ++r) {
        int gr = row0 + wm * 32 + mt * 16 + ((lane >> 4) << 3) + r;
        int gc = col0 + wn * 64 + nt * 16 + (lane & 15);
        float val = acc[mt][nt][r];
        if (bias) val += bias[gc];
        if (do_relu) val = fmaxf(val, 0.f);
        if (resid) val += resid[(size_t)gr * N + gc];
        C[(size_t)gr * N + gc] = val;
      }
    }
  }
}

// ---------------- LSH hashing: buckets[bh][h*T+t] ----------------------------
__global__ __launch_bounds__(256) void hash_kernel(
    const float* __restrict__ qk, const float* __restrict__ rot,
    unsigned char* __restrict__ buckets) {
  int g  = blockIdx.x * 256 + threadIdx.x;   // 32*4096 threads
  int bh = g >> 12;
  int t  = g & (T_SZ - 1);
  int b  = bh >> 3, h = bh & 7;
  const float4* row4 = (const float4*)(qk + ((size_t)b * T_SZ + t) * D_SZ + h * DH_SZ);
  float q[DH_SZ];
#pragma unroll
  for (int f = 0; f < 16; ++f) {
    float4 x = row4[f];
    q[4 * f + 0] = x.x; q[4 * f + 1] = x.y; q[4 * f + 2] = x.z; q[4 * f + 3] = x.w;
  }

  for (int hh = 0; hh < NHASH; ++hh) {
    float best = -3.4e38f;
    int bi = 0;
    for (int i = 0; i < 32; ++i) {
      float d = 0.f;
#pragma unroll
      for (int f = 0; f < DH_SZ; ++f) d += q[f] * rot[f * 128 + hh * 32 + i];
      if (d > best)  { best = d;  bi = i; }
      if (-d > best) { best = -d; bi = i + 32; }
    }
    buckets[(size_t)bh * SORTN + hh * T_SZ + t] = (unsigned char)(bi + hh * NBUCK);
  }
}

// ---------------- stable counting sort per bh --------------------------------
__global__ __launch_bounds__(256) void sort_kernel(
    const unsigned char* __restrict__ buckets,
    int* __restrict__ sticker, int* __restrict__ undo) {
  __shared__ int cnt[256];
  __shared__ int start[256];
  const int tid = threadIdx.x;
  const int bh  = blockIdx.x;
  const unsigned char* bk = buckets + (size_t)bh * SORTN;

  cnt[tid] = 0;
  __syncthreads();
  for (int i = tid; i < SORTN; i += 256) atomicAdd(&cnt[bk[i]], 1);
  __syncthreads();
  if (tid == 0) {
    int s = 0;
    for (int k = 0; k < 256; ++k) { start[k] = s; s += cnt[k]; }
  }
  __syncthreads();
  // thread `tid` owns bucket value `tid`; scan items in order -> stable
  int pos = start[tid];
  for (int i = 0; i < SORTN; ++i) {
    if ((int)bk[i] == tid) {
      sticker[(size_t)bh * SORTN + pos] = i;
      undo[(size_t)bh * SORTN + i]      = pos;
      ++pos;
    }
  }
}

// ---------------- chunked LSH attention (WMMA) -------------------------------
// One block per (bh, chunk). dots = bq@bk^T (64x128) via WMMA, mask, softmax,
// bo = p@bv (64x64) via WMMA. ~57KB LDS.
__global__ __launch_bounds__(256) void lsh_attn_kernel(
    const float* __restrict__ qk, const float* __restrict__ v,
    const int* __restrict__ sticker,
    float* __restrict__ so, float* __restrict__ slog) {
  __shared__ _Float16 sBq[64 * 64];     // [i][e] raw q rows (cur chunk)
  __shared__ _Float16 sBk[128 * 64];    // [j][e] normalized k rows (cur+prev)
  __shared__ _Float16 sBv[128 * 64];    // [j][e] v rows (cur+prev)
  __shared__ _Float16 sD[64 * 128];     // dots f16, later overwritten by p f16
  __shared__ int   sT[128];             // token ids (cur 0..63, prev 64..127)
  __shared__ float sNorm[128];
  __shared__ float sLse[64];

  const int tid = threadIdx.x;
  const int blk = blockIdx.x;
  const int bh  = blk >> 8;
  const int c   = blk & 255;
  const int b   = bh >> 3, h = bh & 7;
  const int pc  = (c + CHUNKS - 1) & 255;

  if (tid < 128) {
    int cc   = (tid < 64) ? c : pc;
    int item = sticker[(size_t)bh * SORTN + cc * BUCKET + (tid & 63)];
    sT[tid]  = item & (T_SZ - 1);       // item % T
  }
  __syncthreads();
  if (tid < 128) {                       // k-row norms (reference: /(||k||+1e-6))
    const float4* r4 =
        (const float4*)(qk + ((size_t)b * T_SZ + sT[tid]) * D_SZ + h * DH_SZ);
    float4 xr[16];
#pragma unroll
    for (int e = 0; e < 16; ++e) xr[e] = r4[e];        // batch loads
    float s = 0.f;
#pragma unroll
    for (int e = 0; e < 16; ++e) {
      float4 x = xr[e];
      s += x.x * x.x + x.y * x.y + x.z * x.z + x.w * x.w;
    }
    sNorm[tid] = sqrtf(s) + 1e-6f;
  }
  __syncthreads();
  {
    // batch all gather loads, then convert + store
    float4 fq[4], fk[8], fv[8];
#pragma unroll
    for (int it = 0; it < 4; ++it) {                   // bq: 64x64
      int l = tid * 4 + it * 1024;
      int i = l >> 6, e = l & 63;
      fq[it] = *(const float4*)(qk + ((size_t)b * T_SZ + sT[i]) * D_SZ + h * DH_SZ + e);
    }
#pragma unroll
    for (int it = 0; it < 8; ++it) {                   // bk/bv: 128x64
      int l = tid * 4 + it * 1024;
      int j = l >> 6, e = l & 63;
      size_t gi = ((size_t)b * T_SZ + sT[j]) * D_SZ + h * DH_SZ + e;
      fk[it] = *(const float4*)(qk + gi);
      fv[it] = *(const float4*)(v + gi);
    }
#pragma unroll
    for (int it = 0; it < 4; ++it) {
      int l = tid * 4 + it * 1024;
      *(v4h*)(sBq + l) = cvt4(fq[it]);
    }
#pragma unroll
    for (int it = 0; it < 8; ++it) {
      int l = tid * 4 + it * 1024;
      int j = l >> 6;
      float inv = 1.0f / sNorm[j];
      float4 k4 = fk[it];
      k4.x *= inv; k4.y *= inv; k4.z *= inv; k4.w *= inv;
      *(v4h*)(sBk + l) = cvt4(k4);
      *(v4h*)(sBv + l) = cvt4(fv[it]);
    }
  }
  __syncthreads();

  // ---- dots = bq @ bk^T * d^-0.5, self-mask -> sD (f16) ----
  {
    const int wid = tid >> 5, lane = tid & 31;
    const int mt = wid & 3;                         // 4 m-tiles
    v16h aLo = frag_a_rowmajor(sBq + (mt * 16) * 64 + 0, 64);
    v16h aHi = frag_a_rowmajor(sBq + (mt * 16) * 64 + 32, 64);
#pragma unroll
    for (int q = 0; q < 4; ++q) {
      int nt = (wid >> 2) * 4 + q;                  // 8 n-tiles / 2 wave groups
      v8f acc = zero8();
      v16h b0 = frag_b_nmajor(sBk + (nt * 16) * 64 + 0, 64);
      v16h b1 = frag_b_nmajor(sBk + (nt * 16) * 64 + 32, 64);
      acc = wmma_f16(aLo, b0, acc);
      acc = wmma_f16(aHi, b1, acc);
      int i0 = mt * 16 + ((lane >> 4) << 3);
      int j  = nt * 16 + (lane & 15);
#pragma unroll
      for (int r = 0; r < 8; ++r) {
        int i = i0 + r;
        float val = acc[r] * 0.125f;                // 64^-0.5
        if (sT[i] == sT[j]) val = -5e4f;            // TOKEN_SELF_ATTN_VALUE
        sD[i * 128 + j] = (_Float16)val;
      }
    }
  }
  __syncthreads();

  // ---- per-row logsumexp, p = exp(dots - lse) written in place (same slot) --
  if (tid < 64) {
    _Float16* drow = sD + tid * 128;
    float m = -3.4e38f;
    for (int j = 0; j < 128; ++j) m = fmaxf(m, (float)drow[j]);
    float s = 0.f;
    for (int j = 0; j < 128; ++j) s += __expf((float)drow[j] - m);
    float lse = m + __logf(s);
    sLse[tid] = lse;
    for (int j = 0; j < 128; ++j) {
      float d = (float)drow[j];
      drow[j] = (_Float16)__expf(d - lse);
    }
  }
  __syncthreads();

  // ---- bo = p @ bv (64x128 @ 128x64) ----
  {
    const int wid = tid >> 5, lane = tid & 31;
#pragma unroll
    for (int q = 0; q < 2; ++q) {
      int tt = wid * 2 + q;                         // 16 tiles / 8 waves
      int mt = tt >> 2, nt = tt & 3;
      v8f acc = zero8();
#pragma unroll
      for (int k0 = 0; k0 < 128; k0 += 32) {
        v16h af = frag_a_rowmajor(sD + (mt * 16) * 128 + k0, 128);
        v16h bf = frag_b_kmajor(sBv + k0 * 64 + nt * 16, 64);
        acc = wmma_f16(af, bf, acc);
      }
      int i0 = mt * 16 + ((lane >> 4) << 3);
      int e  = nt * 16 + (lane & 15);
#pragma unroll
      for (int r = 0; r < 8; ++r) {
        int i = i0 + r;
        so[((size_t)bh * SORTN + c * BUCKET + i) * DH_SZ + e] = acc[r];
      }
    }
  }
  if (tid < 64) slog[(size_t)bh * SORTN + c * BUCKET + tid] = sLse[tid];
}

// ---------------- unsort + combine hash rounds -> (B,T,512) ------------------
__global__ __launch_bounds__(256) void combine_kernel(
    const float* __restrict__ so, const float* __restrict__ slog,
    const int* __restrict__ undo, float* __restrict__ out) {
  int g    = blockIdx.x * 256 + threadIdx.x;   // 32*4096*64 threads
  int e    = g & 63;
  int pair = g >> 6;
  int bh   = pair >> 12;
  int t    = pair & (T_SZ - 1);
  int b    = bh >> 3, h = bh & 7;

  int   pos[NHASH];
  float lg[NHASH];
#pragma unroll
  for (int r = 0; r < NHASH; ++r)
    pos[r] = undo[(size_t)bh * SORTN + r * T_SZ + t];
#pragma unroll
  for (int r = 0; r < NHASH; ++r)
    lg[r]  = slog[(size_t)bh * SORTN + pos[r]];
  float m = lg[0];
#pragma unroll
  for (int r = 1; r < NHASH; ++r) m = fmaxf(m, lg[r]);
  float w[NHASH], s = 0.f;
#pragma unroll
  for (int r = 0; r < NHASH; ++r) { w[r] = __expf(lg[r] - m); s += w[r]; }
  float acc = 0.f;
#pragma unroll
  for (int r = 0; r < NHASH; ++r)
    acc += (w[r] / s) * so[((size_t)bh * SORTN + pos[r]) * DH_SZ + e];

  out[((size_t)b * T_SZ + t) * D_SZ + h * DH_SZ + e] = acc;
}

// ---------------- final LayerNorm --------------------------------------------
__global__ __launch_bounds__(256) void layernorm_kernel(
    const float* __restrict__ y, const float* __restrict__ gamma,
    const float* __restrict__ beta, float* __restrict__ out) {
  __shared__ float red[256];
  const int tid = threadIdx.x;
  const float* row = y + (size_t)blockIdx.x * D_SZ;
  float a = row[tid], c = row[tid + 256];
  red[tid] = a + c;
  __syncthreads();
  for (int st = 128; st > 0; st >>= 1) {
    if (tid < st) red[tid] += red[tid + st];
    __syncthreads();
  }
  float mean = red[0] / (float)D_SZ;
  __syncthreads();
  float d0 = a - mean, d1 = c - mean;
  red[tid] = d0 * d0 + d1 * d1;
  __syncthreads();
  for (int st = 128; st > 0; st >>= 1) {
    if (tid < st) red[tid] += red[tid + st];
    __syncthreads();
  }
  float inv = rsqrtf(red[0] / (float)D_SZ + 1e-6f);
  out[(size_t)blockIdx.x * D_SZ + tid]       = d0 * inv * gamma[tid] + beta[tid];
  out[(size_t)blockIdx.x * D_SZ + tid + 256] = d1 * inv * gamma[tid + 256] + beta[tid + 256];
}

// ---------------- host-side orchestration ------------------------------------
extern "C" void kernel_launch(void* const* d_in, const int* in_sizes, int n_in,
                              void* d_out, int out_size, void* d_ws, size_t ws_size,
                              hipStream_t stream) {
  (void)in_sizes; (void)n_in; (void)out_size; (void)ws_size;
  const float* src   = (const float*)d_in[0];
  const float* Wqk   = (const float*)d_in[1];
  const float* Wv    = (const float*)d_in[2];
  const float* Wout  = (const float*)d_in[3];
  const float* bout  = (const float*)d_in[4];
  const float* W1    = (const float*)d_in[5];
  const float* b1    = (const float*)d_in[6];
  const float* W2    = (const float*)d_in[7];
  const float* b2    = (const float*)d_in[8];
  const float* gamma = (const float*)d_in[9];
  const float* beta  = (const float*)d_in[10];
  const float* rot   = (const float*)d_in[11];
  float* out = (float*)d_out;

  char* ws = (char*)d_ws;
  size_t off = 0;
  auto take = [&](size_t bytes) -> char* {
    char* p = ws + off;
    off += (bytes + 255) & ~(size_t)255;
    return p;
  };
  float*         qkb  = (float*)take((size_t)MTOT * D_SZ * 4);          // 32MB
  float*         vb   = (float*)take((size_t)MTOT * D_SZ * 4);          // 32MB
  unsigned char* bkt  = (unsigned char*)take((size_t)BH_SZ * SORTN);    // 0.5MB
  int*           stk  = (int*)take((size_t)BH_SZ * SORTN * 4);          // 2MB
  int*           und  = (int*)take((size_t)BH_SZ * SORTN * 4);          // 2MB
  float*         so   = (float*)take((size_t)BH_SZ * SORTN * DH_SZ * 4);// 128MB
  float*         slog = (float*)take((size_t)BH_SZ * SORTN * 4);        // 2MB
  float*         ocmb = (float*)take((size_t)MTOT * D_SZ * 4);          // 32MB
  float*         xb   = (float*)take((size_t)MTOT * D_SZ * 4);          // 32MB
  float*         hb   = so;    // reuse: so dead after combine; 16384*2048*4 = 128MB
  float*         yb   = qkb;   // reuse: qk dead after attention

  dim3 blk(256);
  // projections: qk = src@Wqk, v = src@Wv
  gemm_f16wmma<<<dim3(D_SZ / 128, MTOT / 128), blk, 0, stream>>>(
      src, Wqk, nullptr, nullptr, qkb, MTOT, D_SZ, D_SZ, 0);
  gemm_f16wmma<<<dim3(D_SZ / 128, MTOT / 128), blk, 0, stream>>>(
      src, Wv, nullptr, nullptr, vb, MTOT, D_SZ, D_SZ, 0);
  // LSH buckets, stable sort, chunked attention, unsort+combine
  hash_kernel<<<(BH_SZ * T_SZ) / 256, blk, 0, stream>>>(qkb, rot, bkt);
  sort_kernel<<<BH_SZ, blk, 0, stream>>>(bkt, stk, und);
  lsh_attn_kernel<<<BH_SZ * CHUNKS, blk, 0, stream>>>(qkb, vb, stk, so, slog);
  combine_kernel<<<(BH_SZ * T_SZ * DH_SZ) / 256, blk, 0, stream>>>(so, slog, und, ocmb);
  // x = src + (o@Wout + bout)
  gemm_f16wmma<<<dim3(D_SZ / 128, MTOT / 128), blk, 0, stream>>>(
      ocmb, Wout, bout, src, xb, MTOT, D_SZ, D_SZ, 0);
  // h = relu(x@W1 + b1)
  gemm_f16wmma<<<dim3(2048 / 128, MTOT / 128), blk, 0, stream>>>(
      xb, W1, b1, nullptr, hb, MTOT, 2048, D_SZ, 1);
  // y = x + (h@W2 + b2)
  gemm_f16wmma<<<dim3(D_SZ / 128, MTOT / 128), blk, 0, stream>>>(
      hb, W2, b2, xb, yb, MTOT, D_SZ, 2048, 0);
  // out = layernorm(y) * gamma + beta
  layernorm_kernel<<<MTOT, blk, 0, stream>>>(yb, gamma, beta, out);
}
